// MAB_9234179687599
// MI455X (gfx1250) — matrix operand order repaired and checked by reading, not compiled, not run on hardware
//
#include <hip/hip_runtime.h>

// ---------------------------------------------------------------------------
// MAB (multi-head attention block) for MI455X / gfx1250.
// All matmul work goes through v_wmma_f32_16x16x32_bf16 (bf16 in, f32 acc).
// K-tile staging in the attention kernel uses GLOBAL_LOAD_ASYNC_TO_LDS_B64
// (ASYNCcnt path) when the builtin is available.
// B=8, LQ=LK=1024, D=1024, H=16, DH=64.
// ---------------------------------------------------------------------------

typedef __attribute__((ext_vector_type(16))) __bf16 v16bf;
typedef __attribute__((ext_vector_type(8)))  float  v8f;

#define WMMA_BF16(A_, B_, C_) \
  __builtin_amdgcn_wmma_f32_16x16x32_bf16(false, (A_), false, (B_), (short)0, (C_), false, false)

#define NB   8
#define NL   1024
#define ND   1024
#define NH   16
#define NDH  64

#if defined(__gfx1250__) && __has_builtin(__builtin_amdgcn_global_load_async_to_lds_b64)
#define MAB_ASYNC_LDS 1
// Builtin signature (from hipcc diagnostic): param0 is AS1 pointer to a
// GCC-style <2 x i32> vector; param1 the AS3 (LDS) counterpart; then
// imm offset + imm cpol.
typedef int mab_i2 __attribute__((vector_size(8)));
typedef __attribute__((address_space(1))) mab_i2 mab_gi2;  // global int2
typedef __attribute__((address_space(3))) mab_i2 mab_li2;  // LDS int2
#endif

static __device__ __forceinline__ void mab_wait_async() {
#if defined(MAB_ASYNC_LDS)
#if __has_builtin(__builtin_amdgcn_s_wait_asynccnt)
  __builtin_amdgcn_s_wait_asynccnt(0);
#else
  asm volatile("s_wait_asynccnt 0x0" ::: "memory");
#endif
#endif
}

// --- A fragment (16x32 bf16, MxK), p -> element (m=0,k=0), row stride ld ---
// Layout: lanes 0-15 hold M=0..15 with K = kh..kh+15 (contiguous),
// kh = 0 for lanes 0-15, 16 for lanes 16-31.  Contiguous 16B per octet.
static __device__ __forceinline__ v16bf frag_a_ld(const __bf16* __restrict__ p, int ld) {
  const int lane = threadIdx.x & 31;
  const int m    = lane & 15;
  const int kh   = (lane >> 4) << 4;            // 0 or 16
  const __bf16* r = p + (size_t)m * ld;
  v16bf a;
#pragma unroll
  for (int j = 0; j < 8; ++j) {
    a[j]     = r[kh + j];
    a[8 + j] = r[kh + 8 + j];
  }
  return a;
}

// --- B fragment (32x16 bf16, KxN), tile stored N-major: (k,n) at p[n*ld+k] --
// Layout (contiguous k, same as frag_a_ld): lane n holds column n; lanes 0-15 K=0..15, lanes 16-31 K=16..31.
// 16 contiguous bf16 per lane -> ds_load_b128 friendly.
static __device__ __forceinline__ v16bf frag_b_ld(const __bf16* __restrict__ p, int ld) {
  const int lane = threadIdx.x & 31;
  const int n    = lane & 15;
  const int kh   = (lane >> 4) << 4;            // 0 or 16
  const __bf16* c = p + (size_t)n * ld + kh;
  v16bf b;
#pragma unroll
  for (int j = 0; j < 16; ++j) b[j] = c[j];
  return b;
}

// ---------------------------------------------------------------------------
// GEMM: C[8192x1024] = A[8192x1024] @ W[1024x1024] + bias, bf16 WMMA.
// EPI 0: write f32 + bf16 (q projection)
// EPI 1: write bf16 only (k / v projections)
// EPI 2: write f32 = resid + relu(acc + bias)   (output projection)
// Block tile 128x128, 8 waves each computing 64x32 (4x2 WMMA accumulators).
// ---------------------------------------------------------------------------
template <int EPI>
__global__ __launch_bounds__(256) void mab_gemm_kernel(
    const float* __restrict__ A, const float* __restrict__ W,
    const float* __restrict__ bias,
    float* __restrict__ outF, __bf16* __restrict__ outB,
    const float* __restrict__ resid) {
  __shared__ __bf16 As[128][40];     // [m][k], stride 40 (80B, 16B aligned)
  __shared__ __bf16 Bst[128][40];    // [n][k] (transposed W tile)

  const int tid  = threadIdx.x;
  const int row0 = blockIdx.y * 128;
  const int col0 = blockIdx.x * 128;
  const int wid  = tid >> 5;
  const int wr   = (wid & 1) * 64;   // wave row offset in block tile
  const int wc   = (wid >> 1) * 32;  // wave col offset in block tile

  v8f acc[4][2];
#pragma unroll
  for (int i = 0; i < 4; ++i)
#pragma unroll
    for (int j = 0; j < 2; ++j) acc[i][j] = (v8f)0.0f;

  for (int k0 = 0; k0 < ND; k0 += 32) {
    __syncthreads();
    // Stage A (128x32 f32 -> bf16) and W (32x128 f32 -> bf16, transposed).
#pragma unroll
    for (int i = 0; i < 4; ++i) {
      const int e  = tid + i * 256;                 // float4 index 0..1023
      const int ar = e >> 3, ac = (e & 7) * 4;      // A: 8 float4 per row
      const float4 fa = *(const float4*)(A + (size_t)(row0 + ar) * ND + k0 + ac);
      As[ar][ac + 0] = (__bf16)fa.x;
      As[ar][ac + 1] = (__bf16)fa.y;
      As[ar][ac + 2] = (__bf16)fa.z;
      As[ar][ac + 3] = (__bf16)fa.w;
      const int wk = e >> 5, wn = (e & 31) * 4;     // W: 32 float4 per row
      const float4 fw = *(const float4*)(W + (size_t)(k0 + wk) * ND + col0 + wn);
      Bst[wn + 0][wk] = (__bf16)fw.x;
      Bst[wn + 1][wk] = (__bf16)fw.y;
      Bst[wn + 2][wk] = (__bf16)fw.z;
      Bst[wn + 3][wk] = (__bf16)fw.w;
    }
    if (k0 + 32 < ND) {  // prefetch next K slab (global_prefetch_b8)
      __builtin_prefetch(A + (size_t)(row0 + (tid & 127)) * ND + k0 + 32, 0, 1);
      __builtin_prefetch(W + (size_t)(k0 + 32 + (tid & 31)) * ND + col0, 0, 1);
    }
    __syncthreads();

    v16bf af[4], bf2[2];
#pragma unroll
    for (int i = 0; i < 4; ++i) af[i] = frag_a_ld(&As[wr + 16 * i][0], 40);
#pragma unroll
    for (int j = 0; j < 2; ++j) bf2[j] = frag_b_ld(&Bst[wc + 16 * j][0], 40);
#pragma unroll
    for (int i = 0; i < 4; ++i)
#pragma unroll
      for (int j = 0; j < 2; ++j) acc[i][j] = WMMA_BF16(af[i], bf2[j], acc[i][j]);
  }

  // Epilogue. C layout: lane (half,n); VGPR r -> row 8*half + r, col n.
  const int lane = tid & 31;
  const int half = lane >> 4;
  const int n    = lane & 15;
#pragma unroll
  for (int i = 0; i < 4; ++i) {
#pragma unroll
    for (int j = 0; j < 2; ++j) {
      const int col  = col0 + wc + 16 * j + n;
      const float bv = bias[col];
#pragma unroll
      for (int r = 0; r < 8; ++r) {
        const int row    = row0 + wr + 16 * i + half * 8 + r;
        const size_t idx = (size_t)row * ND + col;
        float v = acc[i][j][r] + bv;
        if (EPI == 2) {
          outF[idx] = fmaxf(v, 0.0f) + resid[idx];
        } else if (EPI == 0) {
          outF[idx] = v;
          outB[idx] = (__bf16)v;
        } else {
          outB[idx] = (__bf16)v;
        }
      }
    }
  }
}

// ---------------------------------------------------------------------------
// Flash attention: block = (b, h, 64 queries); 4 waves x 16 query rows.
// Loops over 128-key tiles: S = qk^T/32 + maskbias, online softmax, O += P V.
// K tile staged via async global->LDS DMA (ASYNCcnt); V tile transposed
// through VGPRs.  Writes obuf = q_f32 + softmax(S) V.
// ---------------------------------------------------------------------------
__global__ __launch_bounds__(128) void mab_attn_kernel(
    const __bf16* __restrict__ qbf, const __bf16* __restrict__ kbf,
    const __bf16* __restrict__ vbf, const float* __restrict__ qres,
    const unsigned char* __restrict__ mask, float* __restrict__ obuf) {
  __shared__ __bf16 Ks[128][72];      // [key][d]   (score B-frags: contiguous d)
  __shared__ __bf16 Vst[64][136];     // [d][key]   (AV B-frags: contiguous key)
  __shared__ __bf16 Pls[4][16][136];  // per-wave P tile (C-layout -> A-layout)
  __shared__ float  mb[128];          // additive mask bias per key

  const int tid  = threadIdx.x;
  const int wid  = tid >> 5;
  const int lane = tid & 31;
  const int half = lane >> 4;
  const int n    = lane & 15;
  const int h    = blockIdx.y;
  const int bi   = blockIdx.z;
  const int q0   = blockIdx.x * 64 + wid * 16;

  // q fragments stay in registers for the whole key loop.
  const __bf16* qp = qbf + ((size_t)bi * NL + q0) * ND + h * NDH;
  const v16bf qa0 = frag_a_ld(qp, ND);
  const v16bf qa1 = frag_a_ld(qp + 32, ND);

  v8f oacc[4];
#pragma unroll
  for (int i = 0; i < 4; ++i) oacc[i] = (v8f)0.0f;
  float mrow[8], lrow[8];
#pragma unroll
  for (int r = 0; r < 8; ++r) { mrow[r] = -3.0e38f; lrow[r] = 0.0f; }

  for (int kt = 0; kt < NL; kt += 128) {
    __syncthreads();
    const size_t kvb = ((size_t)bi * NL + kt) * ND + h * NDH;
#pragma unroll
    for (int i = 0; i < 16; ++i) {
      const int e   = tid + i * 128;      // quad index, 128 keys x 16 quads
      const int key = e >> 4;
      const int d4  = (e & 15) * 4;
      const __bf16* gk = kbf + kvb + (size_t)key * ND + d4;
#if defined(MAB_ASYNC_LDS)
      // Per-lane async DMA: global -> LDS, 8 bytes (4 bf16).
      __builtin_amdgcn_global_load_async_to_lds_b64(
          (mab_gi2*)(uintptr_t)gk,
          (mab_li2*)(unsigned)(uintptr_t)(void*)&Ks[key][d4],
          0, 0);
#else
      *(uint2*)&Ks[key][d4] = *(const uint2*)gk;
#endif
      const uint2 rv = *(const uint2*)(vbf + kvb + (size_t)key * ND + d4);
      const __bf16* pv = (const __bf16*)&rv;
      Vst[d4 + 0][key] = pv[0];
      Vst[d4 + 1][key] = pv[1];
      Vst[d4 + 2][key] = pv[2];
      Vst[d4 + 3][key] = pv[3];
    }
    mb[tid] = mask[(size_t)bi * NL + kt + tid] ? 0.0f : -1.0e10f;
    mab_wait_async();   // drain ASYNCcnt before making the tile visible
    __syncthreads();

    // Scores: 8 subtiles of 16 keys, K-dim = head dim (2 WMMA steps).
    v8f S[8];
#pragma unroll
    for (int ct = 0; ct < 8; ++ct) {
      const v16bf b0 = frag_b_ld(&Ks[ct * 16][0], 72);
      const v16bf b1 = frag_b_ld(&Ks[ct * 16][32], 72);
      v8f s = (v8f)0.0f;
      s = WMMA_BF16(qa0, b0, s);
      s = WMMA_BF16(qa1, b1, s);
      S[ct] = s;
    }
    float mbl[8];
#pragma unroll
    for (int ct = 0; ct < 8; ++ct) mbl[ct] = mb[ct * 16 + n];

    // Online softmax; row reductions across the 16-lane half via shfl_xor.
#pragma unroll
    for (int r = 0; r < 8; ++r) {
      float tmax = -3.0e38f;
#pragma unroll
      for (int ct = 0; ct < 8; ++ct) {
        const float s = S[ct][r] * 0.03125f + mbl[ct];  // 1/sqrt(1024)
        S[ct][r] = s;
        tmax = fmaxf(tmax, s);
      }
#pragma unroll
      for (int off = 8; off > 0; off >>= 1) tmax = fmaxf(tmax, __shfl_xor(tmax, off, 32));
      const float nm      = fmaxf(mrow[r], tmax);
      const float rescale = __expf(mrow[r] - nm);
      mrow[r] = nm;
      float rs = 0.0f;
#pragma unroll
      for (int ct = 0; ct < 8; ++ct) {
        const float p = __expf(S[ct][r] - nm);
        S[ct][r] = p;
        rs += p;
      }
#pragma unroll
      for (int off = 8; off > 0; off >>= 1) rs += __shfl_xor(rs, off, 32);
      lrow[r] = lrow[r] * rescale + rs;
#pragma unroll
      for (int i = 0; i < 4; ++i) oacc[i][r] *= rescale;
    }

    // P: C-layout -> LDS (per-wave; same-wave DS ops stay in order).
#pragma unroll
    for (int r = 0; r < 8; ++r) {
      const int row = half * 8 + r;
#pragma unroll
      for (int ct = 0; ct < 8; ++ct)
        Pls[wid][row][ct * 16 + n] = (__bf16)S[ct][r];
    }

    // O += P @ V : K-dim = 128 keys (4 WMMA steps), 4 head-dim subtiles.
#pragma unroll
    for (int ks = 0; ks < 4; ++ks) {
      const v16bf pa = frag_a_ld(&Pls[wid][0][ks * 32], 136);
#pragma unroll
      for (int ct = 0; ct < 4; ++ct) {
        const v16bf vb = frag_b_ld(&Vst[ct * 16][ks * 32], 136);
        oacc[ct] = WMMA_BF16(pa, vb, oacc[ct]);
      }
    }
  }

  // o = q_f32 + O / l
#pragma unroll
  for (int r = 0; r < 8; ++r) {
    const int row   = q0 + half * 8 + r;
    const float inv = 1.0f / lrow[r];
#pragma unroll
    for (int ct = 0; ct < 4; ++ct) {
      const int col    = h * NDH + ct * 16 + n;
      const size_t idx = ((size_t)bi * NL + row) * ND + col;
      obuf[idx] = qres[idx] + oacc[ct][r] * inv;
    }
  }
}

// ---------------------------------------------------------------------------
// LayerNorm over rows of 1024, one block per row.
// ---------------------------------------------------------------------------
__global__ __launch_bounds__(256) void mab_ln_kernel(
    const float* __restrict__ x, const float* __restrict__ g,
    const float* __restrict__ be, float* __restrict__ y) {
  const int row = blockIdx.x;
  const int tid = threadIdx.x;
  const float* xr = x + (size_t)row * ND;
  const float4 v = *(const float4*)(xr + tid * 4);
  float s  = v.x + v.y + v.z + v.w;
  float ss = v.x * v.x + v.y * v.y + v.z * v.z + v.w * v.w;
#pragma unroll
  for (int off = 16; off > 0; off >>= 1) {
    s  += __shfl_xor(s, off, 32);
    ss += __shfl_xor(ss, off, 32);
  }
  __shared__ float rs[8], rss[8];
  const int wid = tid >> 5, lane = tid & 31;
  if (lane == 0) { rs[wid] = s; rss[wid] = ss; }
  __syncthreads();
  float S = 0.0f, SS = 0.0f;
#pragma unroll
  for (int i = 0; i < 8; ++i) { S += rs[i]; SS += rss[i]; }
  const float mean = S * (1.0f / ND);
  const float var  = SS * (1.0f / ND) - mean * mean;
  const float rstd = rsqrtf(var + 1e-5f);
  const int c = tid * 4;
  const float4 gv = *(const float4*)(g + c);
  const float4 bv = *(const float4*)(be + c);
  float4 o;
  o.x = (v.x - mean) * rstd * gv.x + bv.x;
  o.y = (v.y - mean) * rstd * gv.y + bv.y;
  o.z = (v.z - mean) * rstd * gv.z + bv.z;
  o.w = (v.w - mean) * rstd * gv.w + bv.w;
  *(float4*)(y + (size_t)row * ND + c) = o;
}

// ---------------------------------------------------------------------------
extern "C" void kernel_launch(void* const* d_in, const int* in_sizes, int n_in,
                              void* d_out, int out_size, void* d_ws, size_t ws_size,
                              hipStream_t stream) {
  (void)in_sizes; (void)n_in; (void)out_size; (void)ws_size;
  const float* Q    = (const float*)d_in[0];
  const float* Kin  = (const float*)d_in[1];
  const unsigned char* mask = (const unsigned char*)d_in[2];  // jax bool, 1 byte
  const float* Wq = (const float*)d_in[3];
  const float* bq = (const float*)d_in[4];
  const float* Wk = (const float*)d_in[5];
  const float* bk = (const float*)d_in[6];
  const float* Wv = (const float*)d_in[7];
  const float* bv = (const float*)d_in[8];
  const float* Wo = (const float*)d_in[9];
  const float* bo = (const float*)d_in[10];
  const float* g0 = (const float*)d_in[11];
  const float* b0 = (const float*)d_in[12];
  const float* g1 = (const float*)d_in[13];
  const float* b1 = (const float*)d_in[14];

  char* ws = (char*)d_ws;
  const size_t F32 = (size_t)NB * NL * ND * sizeof(float);   // 32 MB
  const size_t BF  = (size_t)NB * NL * ND * sizeof(__bf16);  // 16 MB
  float*  qf32 = (float*)ws;                 // q proj f32 (later reused as x0)
  __bf16* qbf  = (__bf16*)(ws + F32);
  __bf16* kbf  = (__bf16*)(ws + F32 + BF);
  __bf16* vbf  = (__bf16*)(ws + F32 + 2 * BF);
  float*  obuf = (float*)(ws + F32 + 3 * BF);  // attn out / gemm2 out

  const dim3 gg(ND / 128, (NB * NL) / 128);   // (8, 64)
  mab_gemm_kernel<0><<<gg, 256, 0, stream>>>(Q,   Wq, bq, qf32,  qbf, nullptr);
  mab_gemm_kernel<1><<<gg, 256, 0, stream>>>(Kin, Wk, bk, nullptr, kbf, nullptr);
  mab_gemm_kernel<1><<<gg, 256, 0, stream>>>(Kin, Wv, bv, nullptr, vbf, nullptr);

  mab_attn_kernel<<<dim3(NL / 64, NH, NB), 128, 0, stream>>>(qbf, kbf, vbf, qf32, mask, obuf);

  mab_ln_kernel<<<NB * NL, 256, 0, stream>>>(obuf, g0, b0, qf32);      // x0 = LN0(o)
  mab_gemm_kernel<2><<<gg, 256, 0, stream>>>(qf32, Wo, bo, obuf, nullptr, qf32);
  mab_ln_kernel<<<NB * NL, 256, 0, stream>>>(obuf, g1, b1, (float*)d_out);
}
